// PN2Dense_60043642798276
// MI455X (gfx1250) — matrix operand order repaired and checked by reading, not compiled, not run on hardware
//
#include <hip/hip_runtime.h>
#include <cstdint>

typedef _Float16 h16;
typedef __attribute__((ext_vector_type(16))) _Float16 v16h;
typedef __attribute__((ext_vector_type(8)))  float    v8f;

#define B_  16
#define N_  4096
#define N1_ 819
#define N2_ 204

// ---------------------------------------------------------------------------
// WMMA helpers (CDNA5 wave32, V_WMMA_F32_16X16X32_F16)
// A frag (16x32 f16): lane<16 -> row M=lane, K 0..15; lane>=16 -> K 16..31
// B frag (32x16 f16): lane -> col N=lane&15; lane<16 -> K 0..15; lane>=16 -> K 16..31
// C/D (16x16 f32): vgpr j, lane l -> row m0 + j + (l>=16?8:0), col n0 + (l&15)
// ---------------------------------------------------------------------------
__device__ __forceinline__ v8f wmma_f32_f16(v16h a, v16h b, v8f c) {
  return __builtin_amdgcn_wmma_f32_16x16x32_f16(false, a, false, b, (short)0, c,
                                                false, false);
}

__device__ __forceinline__ v16h load_a_frag(const h16* A, int lda, int m0, int kb) {
  const int lane = threadIdx.x & 31;
  const int row  = m0 + (lane & 15);
  const int koff = (lane >> 4) << 4;
  const h16* p0 = A + (size_t)row * lda + kb + koff;
  v16h r;
#pragma unroll
  for (int i = 0; i < 8; ++i) r[i] = p0[i];
#pragma unroll
  for (int i = 0; i < 8; ++i) r[8 + i] = p0[8 + i];
  return r;
}

__device__ __forceinline__ v16h load_b_frag(const h16* WT, int ldw, int n0, int kb) {
  const int lane = threadIdx.x & 31;
  const int col  = n0 + (lane & 15);
  const int klo  = (lane >> 4) << 4;
  const h16* p = WT + (size_t)col * ldw + kb + klo;
  v16h r;
#pragma unroll
  for (int i = 0; i < 16; ++i) r[i] = p[i];
  return r;
}

// act: 0 = none, 1 = relu, 2 = leaky_relu(0.01)
__device__ __forceinline__ void store_c_tile(v8f c, h16* O, int ldo, int m0, int n0,
                                             const float* bias, int nact, int act) {
  const int lane = threadIdx.x & 31;
  const int col  = n0 + (lane & 15);
  const float bv = (col < nact) ? bias[col] : 0.0f;
  const int rbase = m0 + ((lane >> 4) << 3);
#pragma unroll
  for (int j = 0; j < 8; ++j) {
    float v = c[j] + bv;
    if (act == 1)      v = fmaxf(v, 0.0f);
    else if (act == 2) v = (v > 0.0f) ? v : 0.01f * v;
    O[(size_t)(rbase + j) * ldo + col] = (h16)v;
  }
}

// ---------------------------------------------------------------------------
// Register-tiled GEMM: each wave computes a 32x64 output tile (2x4 WMMA frags,
// 8 v_wmma per K step, A frags reused 4x / B frags reused 2x).
// Requires Rp % 32 == 0 (rows padded to 64 by host) and Np % 64 == 0.
// ---------------------------------------------------------------------------
__global__ __launch_bounds__(256) void gemm_tiled_kernel(
    const h16* __restrict__ A, const h16* __restrict__ WT,
    const float* __restrict__ bias, h16* __restrict__ O,
    int Rp, int Kp, int Np, int Nact, int act) {
  const int wave  = threadIdx.x >> 5;
  const int tile  = blockIdx.x * 8 + wave;
  const int ntb   = Np >> 6;                // 64-wide N blocks
  const int total = (Rp >> 5) * ntb;        // 32-row M blocks
  if (tile >= total) return;                // wave-uniform
  const int m0 = (tile / ntb) << 5;
  const int n0 = (tile % ntb) << 6;
  v8f acc[2][4] = {};
  for (int kb = 0; kb < Kp; kb += 32) {
    if (kb + 32 < Kp)                       // prefetch next weight K-slice
      __builtin_prefetch(WT + (size_t)n0 * Kp + kb + 32, 0, 3);
    v16h a[2], b[4];
#pragma unroll
    for (int i = 0; i < 2; ++i) a[i] = load_a_frag(A, Kp, m0 + 16 * i, kb);
#pragma unroll
    for (int j = 0; j < 4; ++j) b[j] = load_b_frag(WT, Kp, n0 + 16 * j, kb);
#pragma unroll
    for (int i = 0; i < 2; ++i)
#pragma unroll
      for (int j = 0; j < 4; ++j)
        acc[i][j] = wmma_f32_f16(a[i], b[j], acc[i][j]);
  }
#pragma unroll
  for (int i = 0; i < 2; ++i)
#pragma unroll
    for (int j = 0; j < 4; ++j)
      store_c_tile(acc[i][j], O, Np, m0 + 16 * i, n0 + 16 * j, bias, Nact, act);
}

// Fallback single-tile GEMM (used for lin3, Np=16)
__global__ __launch_bounds__(256) void gemm_relu_kernel(
    const h16* __restrict__ A, const h16* __restrict__ WT,
    const float* __restrict__ bias, h16* __restrict__ O,
    int R, int Kp, int Np, int Nact, int act) {
  const int wave  = threadIdx.x >> 5;
  const int tile  = blockIdx.x * 8 + wave;
  const int nt    = Np >> 4;
  const int total = (R >> 4) * nt;
  if (tile >= total) return;               // wave-uniform
  const int m0 = (tile / nt) << 4;
  const int n0 = (tile % nt) << 4;
  v8f c = {};
  for (int kb = 0; kb < Kp; kb += 32) {
    v16h a = load_a_frag(A, Kp, m0, kb);
    v16h b = load_b_frag(WT, Kp, n0, kb);
    c = wmma_f32_f16(a, b, c);
  }
  store_c_tile(c, O, Np, m0, n0, bias, Nact, act);
}

// 64-row LDS GEMM used inside fused SA modules (8 waves, tiles = 4 * NP/16)
template <int KP, int NP>
__device__ __forceinline__ void gemm_tiles_lds(const h16* A, const h16* WT,
                                               const float* bias, h16* O, int act) {
  const int wave = threadIdx.x >> 5;
  constexpr int NT = NP >> 4;
  for (int t = wave; t < 4 * NT; t += 8) {
    const int m0 = (t / NT) << 4;
    const int n0 = (t % NT) << 4;
    v8f c = {};
#pragma unroll
    for (int kb = 0; kb < KP; kb += 32) {
      v16h a = load_a_frag(A, KP, m0, kb);
      v16h b = load_b_frag(WT, KP, n0, kb);
      c = wmma_f32_f16(a, b, c);
    }
    store_c_tile(c, O, NP, m0, n0, bias, NP, act);
  }
}

// ---------------------------------------------------------------------------
// Fused SA module: gather [feat, relpos] -> MLP(3 layers, WMMA in LDS) -> masked max
// One workgroup per center; 64 neighbor rows.
// ---------------------------------------------------------------------------
template <int CIN, int C1, int C2, int C3, int FCH>
__global__ __launch_bounds__(256) void sa_fused_kernel(
    const float* __restrict__ srcpos, const h16* __restrict__ srcfeat,
    const float* __restrict__ cpos, const int* __restrict__ nidx,
    const int* __restrict__ ncnt,
    const h16* __restrict__ w1, const float* __restrict__ b1,
    const h16* __restrict__ w2, const float* __restrict__ b2,
    const h16* __restrict__ w3, const float* __restrict__ b3,
    h16* __restrict__ xout) {
  constexpr int CA = (CIN > C2) ? CIN : C2;
  constexpr int CB = (C1 > C3) ? C1 : C3;
  __shared__ h16 bufA[64 * CA];
  __shared__ h16 bufB[64 * CB];
  const int c   = blockIdx.x;
  const int tid = threadIdx.x;
  const int cnt = ncnt[c];
  const float cx = cpos[c * 3 + 0], cy = cpos[c * 3 + 1], cz = cpos[c * 3 + 2];
  // gather input rows (zero padded channels / invalid rows)
  for (int e = tid; e < 64 * CIN; e += 256) {
    const int r = e / CIN, ch = e % CIN;
    h16 v = (h16)0.0f;
    if (r < cnt) {
      const int j = nidx[(size_t)c * 64 + r];
      if (ch < FCH) {
        v = srcfeat[(size_t)j * FCH + ch];
      } else if (ch < FCH + 3) {
        const int d = ch - FCH;
        const float pv = srcpos[(size_t)j * 3 + d] - (d == 0 ? cx : (d == 1 ? cy : cz));
        v = (h16)pv;
      }
    }
    bufA[e] = v;
  }
  __syncthreads();
  gemm_tiles_lds<CIN, C1>(bufA, w1, b1, bufB, 1);
  __syncthreads();
  gemm_tiles_lds<C1, C2>(bufB, w2, b2, bufA, 1);
  __syncthreads();
  gemm_tiles_lds<C2, C3>(bufA, w3, b3, bufB, 0);
  __syncthreads();
  // masked max over valid neighbor rows
  for (int ch = tid; ch < C3; ch += 256) {
    float m = -3.0e38f;
    for (int r = 0; r < cnt; ++r) m = fmaxf(m, (float)bufB[r * C3 + ch]);
    xout[(size_t)c * C3 + ch] = (h16)m;
  }
}

// ---------------------------------------------------------------------------
// Farthest point sampling, one workgroup per cloud, deterministic (tie->lowest idx)
// ---------------------------------------------------------------------------
__global__ __launch_bounds__(1024) void fps_kernel(const float* __restrict__ pos,
                                                   int Nsrc, int Nsel,
                                                   int* __restrict__ outIdx) {
  const int b = blockIdx.x, tid = threadIdx.x;
  const float* p = pos + (size_t)b * Nsrc * 3;
  float px[4], py[4], pz[4], dmin[4];
  bool have[4];
#pragma unroll
  for (int q = 0; q < 4; ++q) {
    const int i = tid + q * 1024;
    have[q] = (i < Nsrc);
    if (have[q]) { px[q] = p[i * 3]; py[q] = p[i * 3 + 1]; pz[q] = p[i * 3 + 2]; }
    else         { px[q] = py[q] = pz[q] = 0.f; }
    dmin[q] = 3.4e38f;
  }
  __shared__ float sval[1024];
  __shared__ int   sidx[1024];
  __shared__ float lx, ly, lz;
  if (tid == 0) { outIdx[(size_t)b * Nsel] = 0; lx = p[0]; ly = p[1]; lz = p[2]; }
  __syncthreads();
  for (int it = 1; it < Nsel; ++it) {
    float bestv = -1.0f; int besti = 0x7fffffff;
#pragma unroll
    for (int q = 0; q < 4; ++q) {
      if (have[q]) {
        const float dx = px[q] - lx, dy = py[q] - ly, dz = pz[q] - lz;
        const float d = dx * dx + dy * dy + dz * dz;
        dmin[q] = fminf(dmin[q], d);
        if (dmin[q] > bestv) { bestv = dmin[q]; besti = tid + q * 1024; }
      }
    }
    sval[tid] = bestv; sidx[tid] = besti;
    __syncthreads();
    for (int s = 512; s > 0; s >>= 1) {
      if (tid < s) {
        if (sval[tid + s] > sval[tid] ||
            (sval[tid + s] == sval[tid] && sidx[tid + s] < sidx[tid])) {
          sval[tid] = sval[tid + s]; sidx[tid] = sidx[tid + s];
        }
      }
      __syncthreads();
    }
    if (tid == 0) {
      const int ni = sidx[0];
      outIdx[(size_t)b * Nsel + it] = ni;
      lx = p[ni * 3]; ly = p[ni * 3 + 1]; lz = p[ni * 3 + 2];
    }
    __syncthreads();
  }
}

__global__ void gather_pos_kernel(const float* __restrict__ src,
                                  const int* __restrict__ idx,
                                  float* __restrict__ dst,
                                  int Msel, int NsrcStride, int total) {
  const int i = blockIdx.x * 256 + threadIdx.x;
  if (i >= total) return;
  const int b = i / Msel;
  const int j = idx[i];
  const float* s = src + ((size_t)b * NsrcStride + j) * 3;
  dst[i * 3 + 0] = s[0]; dst[i * 3 + 1] = s[1]; dst[i * 3 + 2] = s[2];
}

// ---------------------------------------------------------------------------
// K-nearest within radius: binary search on distance threshold for the 64th
// smallest d2, then deterministic index-ordered compaction (no atomics).
// ---------------------------------------------------------------------------
__global__ __launch_bounds__(256) void radius_knn_kernel(
    const float* __restrict__ cpos, const float* __restrict__ spos,
    int M, int Nsrc, float r2, int* __restrict__ nidx, int* __restrict__ ncnt) {
  __shared__ float d2s[4096];
  __shared__ int cnts[256];
  __shared__ int offs[257];
  __shared__ int scnt;
  const int c = blockIdx.x, tid = threadIdx.x;
  const int b = c / M;
  const float* sp = spos + (size_t)b * Nsrc * 3;
  const float cx = cpos[c * 3], cy = cpos[c * 3 + 1], cz = cpos[c * 3 + 2];
  const int per = (Nsrc + 255) >> 8;
  const int beg = tid * per;
  const int end = min(Nsrc, beg + per);
  for (int i = beg; i < end; ++i) {
    const float dx = sp[i * 3] - cx, dy = sp[i * 3 + 1] - cy, dz = sp[i * 3 + 2] - cz;
    d2s[i] = dx * dx + dy * dy + dz * dz;
  }
  __syncthreads();
  int cl = 0;
  for (int i = beg; i < end; ++i) if (d2s[i] <= r2) cl++;
  cnts[tid] = cl;
  __syncthreads();
  if (tid == 0) { int s = 0; for (int k = 0; k < 256; ++k) s += cnts[k]; scnt = s; }
  __syncthreads();
  const int target = min(64, scnt);
  float lo = 0.f, hi = r2;
  for (int it = 0; it < 24; ++it) {
    const float mid = 0.5f * (lo + hi);
    int cc = 0;
    for (int i = beg; i < end; ++i) if (d2s[i] <= mid) cc++;
    cnts[tid] = cc;
    __syncthreads();
    if (tid == 0) { int s = 0; for (int k = 0; k < 256; ++k) s += cnts[k]; scnt = s; }
    __syncthreads();
    if (scnt >= target) hi = mid; else lo = mid;
  }
  const float thr = hi;
  int myc = 0;
  for (int i = beg; i < end; ++i) if (d2s[i] <= thr) myc++;
  cnts[tid] = myc;
  __syncthreads();
  if (tid == 0) {
    int s = 0;
    for (int k = 0; k < 256; ++k) { offs[k] = s; s += cnts[k]; }
    offs[256] = s;
  }
  __syncthreads();
  int o = offs[tid];
  for (int i = beg; i < end; ++i)
    if (d2s[i] <= thr) { if (o < 64) nidx[(size_t)c * 64 + o] = b * Nsrc + i; o++; }
  if (tid == 0) ncnt[c] = min(64, offs[256]);
}

// 3-NN for knn_interpolate (1 thread per dst point; ties -> first index)
__global__ void knn3_kernel(const float* __restrict__ dpos,
                            const float* __restrict__ spos,
                            int Ntot, int Dd, int Ns,
                            int* __restrict__ oidx, float* __restrict__ ow) {
  const int i = blockIdx.x * 256 + threadIdx.x;
  if (i >= Ntot) return;
  const int b = i / Dd;
  const float* sp = spos + (size_t)b * Ns * 3;
  const float px = dpos[i * 3], py = dpos[i * 3 + 1], pz = dpos[i * 3 + 2];
  float d0 = 3e38f, d1 = 3e38f, d2v = 3e38f;
  int i0 = 0, i1 = 0, i2 = 0;
  for (int j = 0; j < Ns; ++j) {
    const float dx = sp[j * 3] - px, dy = sp[j * 3 + 1] - py, dz = sp[j * 3 + 2] - pz;
    const float d = dx * dx + dy * dy + dz * dz;
    if (d < d0)       { d2v = d1; i2 = i1; d1 = d0; i1 = i0; d0 = d; i0 = j; }
    else if (d < d1)  { d2v = d1; i2 = i1; d1 = d;  i1 = j; }
    else if (d < d2v) { d2v = d;  i2 = j; }
  }
  const float w0 = 1.f / fmaxf(d0, 1e-16f);
  const float w1 = 1.f / fmaxf(d1, 1e-16f);
  const float w2 = 1.f / fmaxf(d2v, 1e-16f);
  const float s = w0 + w1 + w2;
  ow[i * 3 + 0] = w0 / s; ow[i * 3 + 1] = w1 / s; ow[i * 3 + 2] = w2 / s;
  oidx[i * 3 + 0] = b * Ns + i0; oidx[i * 3 + 1] = b * Ns + i1; oidx[i * 3 + 2] = b * Ns + i2;
}

// ----------------------------- small utility kernels -----------------------
__global__ void f2h_kernel(const float* __restrict__ x, h16* __restrict__ o, int n) {
  const int i = blockIdx.x * 256 + threadIdx.x;
  if (i < n) o[i] = (h16)x[i];
}

// W fp32 [Cin][Cout] -> WT half [Npad][Kpad] (transposed, zero padded)
__global__ void prep_wt_kernel(const float* __restrict__ W, h16* __restrict__ WT,
                               int Cin, int Cout, int Kp, int Np) {
  const int i = blockIdx.x * 256 + threadIdx.x;
  if (i >= Np * Kp) return;
  const int n = i / Kp, k = i % Kp;
  const float v = (n < Cout && k < Cin) ? W[(size_t)k * Cout + n] : 0.f;
  WT[i] = (h16)v;
}

__global__ void build_gsaA_kernel(const h16* __restrict__ x2, const float* __restrict__ pos2,
                                  h16* __restrict__ A, int total) {
  const int e = blockIdx.x * 256 + threadIdx.x;
  if (e >= total) return;
  const int i = e / 288, ch = e % 288;
  h16 v = (h16)0.f;
  if (ch < 256) v = x2[(size_t)i * 256 + ch];
  else if (ch < 259) v = (h16)pos2[(size_t)i * 3 + (ch - 256)];
  A[e] = v;
}

__global__ void gmax_kernel(const h16* __restrict__ t3, h16* __restrict__ g) {
  const int b = blockIdx.x;
  for (int ch = threadIdx.x; ch < 1024; ch += 256) {
    float m = -3e38f;
    for (int r = 0; r < N2_; ++r)
      m = fmaxf(m, (float)t3[((size_t)b * N2_ + r) * 1024 + ch]);
    g[(size_t)b * 1024 + ch] = (h16)m;
  }
}

__global__ void build_fp3A_kernel(const h16* __restrict__ g, const h16* __restrict__ x2,
                                  h16* __restrict__ A, int total) {
  const int e = blockIdx.x * 256 + threadIdx.x;
  if (e >= total) return;
  const int i = e / 1280, ch = e % 1280;
  const int b = i / N2_;
  A[e] = (ch < 1024) ? g[(size_t)b * 1024 + ch] : x2[(size_t)i * 256 + (ch - 1024)];
}

__global__ void build_fp2A_kernel(const h16* __restrict__ h3, const int* __restrict__ idx3,
                                  const float* __restrict__ w3, const h16* __restrict__ x1,
                                  h16* __restrict__ A, int total) {
  const int e = blockIdx.x * 256 + threadIdx.x;
  if (e >= total) return;
  const int i = e / 384, ch = e % 384;
  h16 v;
  if (ch < 256) {
    float a = 0.f;
#pragma unroll
    for (int t = 0; t < 3; ++t)
      a += w3[i * 3 + t] * (float)h3[(size_t)idx3[i * 3 + t] * 256 + ch];
    v = (h16)a;
  } else {
    v = x1[(size_t)i * 128 + (ch - 256)];
  }
  A[e] = v;
}

__global__ void build_fp1A_kernel(const h16* __restrict__ h2, const int* __restrict__ idx3,
                                  const float* __restrict__ w3, const h16* __restrict__ xh,
                                  h16* __restrict__ A, int total) {
  const int e = blockIdx.x * 256 + threadIdx.x;
  if (e >= total) return;
  const int i = e / 160, ch = e % 160;
  h16 v = (h16)0.f;
  if (ch < 128) {
    float a = 0.f;
#pragma unroll
    for (int t = 0; t < 3; ++t)
      a += w3[i * 3 + t] * (float)h2[(size_t)idx3[i * 3 + t] * 128 + ch];
    v = (h16)a;
  } else if (ch < 131) {
    v = xh[(size_t)i * 3 + (ch - 128)];
  }
  A[e] = v;
}

__global__ void out_kernel(const h16* __restrict__ l3, float* __restrict__ out, int total) {
  const int e = blockIdx.x * 256 + threadIdx.x;
  if (e >= total) return;
  const int i = e / 3, c = e % 3;
  out[e] = (float)l3[(size_t)i * 16 + c];
}

// ---------------------------------------------------------------------------
// Host orchestration
// ---------------------------------------------------------------------------
struct LayerDef { int wi, bi, Cin, Cout, Kp, Np; };
// param leaf order (sorted keys): fp1, fp2, fp3, gsa, lin1, lin2, lin3, sa1, sa2
static const LayerDef LDEFS[19] = {
  {0, 1, 131, 128, 160, 128}, {2, 3, 128, 128, 128, 128}, {4, 5, 128, 128, 128, 128}, // fp1
  {6, 7, 384, 256, 384, 256}, {8, 9, 256, 128, 256, 128},                             // fp2
  {10, 11, 1280, 256, 1280, 256}, {12, 13, 256, 256, 256, 256},                       // fp3
  {14, 15, 259, 256, 288, 256}, {16, 17, 256, 512, 256, 512}, {18, 19, 512, 1024, 512, 1024}, // gsa
  {20, 21, 128, 128, 128, 128}, {22, 23, 128, 128, 128, 128}, {24, 25, 128, 3, 128, 16},      // lin1-3
  {26, 27, 6, 64, 32, 64}, {28, 29, 64, 64, 64, 64}, {30, 31, 64, 128, 64, 128},              // sa1
  {32, 33, 131, 128, 160, 128}, {34, 35, 128, 128, 128, 128}, {36, 37, 128, 256, 128, 256},   // sa2
};

static inline int rpad64(int r) { return (r + 63) & ~63; }

extern "C" void kernel_launch(void* const* d_in, const int* in_sizes, int n_in,
                              void* d_out, int out_size, void* d_ws, size_t ws_size,
                              hipStream_t stream) {
  (void)out_size; (void)ws_size;
  // input-order disambiguation: insertion order (x,pos,batch,params) vs
  // fully-sorted flatten (batch, params..., pos, x). batch has 65536 elems.
  int pbase, ix, ipos;
  if (n_in >= 41 && in_sizes[0] == B_ * N_) { pbase = 1; ipos = 39; ix = 40; }
  else                                      { ix = 0; ipos = 1; pbase = 3; }
  const float* x   = (const float*)d_in[ix];
  const float* pos = (const float*)d_in[ipos];
  auto P = [&](int rel) { return (const float*)d_in[pbase + rel]; };

  uint8_t* base = (uint8_t*)d_ws;
  size_t off = 0;
  auto alloc = [&](size_t bytes) -> void* {
    void* p = base + off;
    off = (off + bytes + 255) & ~(size_t)255;
    return p;
  };

  const int R1 = B_ * N1_;            // 13104 (padded to 13120 for tiled GEMMs)
  const int R1p = rpad64(R1);
  const int R2 = B_ * N2_;            // 3264 (already % 64)
  const int R0 = B_ * N_;             // 65536

  h16* xh = (h16*)alloc((size_t)R0 * 3 * 2);
  h16* wt[19];
  for (int l = 0; l < 19; ++l) wt[l] = (h16*)alloc((size_t)LDEFS[l].Np * LDEFS[l].Kp * 2);
  float* pos1  = (float*)alloc((size_t)R1 * 3 * 4);
  float* pos2  = (float*)alloc((size_t)R2 * 3 * 4);
  int*   idx1  = (int*)alloc((size_t)R1 * 4);
  int*   idx2  = (int*)alloc((size_t)R2 * 4);
  int*   nidx1 = (int*)alloc((size_t)R1 * 64 * 4);
  int*   ncnt1 = (int*)alloc((size_t)R1 * 4);
  int*   nidx2 = (int*)alloc((size_t)R2 * 64 * 4);
  int*   ncnt2 = (int*)alloc((size_t)R2 * 4);
  h16*   x1h   = (h16*)alloc((size_t)R1 * 128 * 2);
  h16*   x2h   = (h16*)alloc((size_t)R2 * 256 * 2);
  h16*   gsaA  = (h16*)alloc((size_t)R2 * 288 * 2);
  h16*   gt1   = (h16*)alloc((size_t)R2 * 256 * 2);
  h16*   gt2   = (h16*)alloc((size_t)R2 * 512 * 2);
  h16*   gt3   = (h16*)alloc((size_t)R2 * 1024 * 2);
  h16*   gpool = (h16*)alloc((size_t)B_ * 1024 * 2);
  h16*   fp3A  = (h16*)alloc((size_t)R2 * 1280 * 2);
  h16*   f3t   = (h16*)alloc((size_t)R2 * 256 * 2);
  h16*   h3    = (h16*)alloc((size_t)R2 * 256 * 2);
  int*   idx3a = (int*)alloc((size_t)R1 * 3 * 4);
  float* w3a   = (float*)alloc((size_t)R1 * 3 * 4);
  h16*   fp2A  = (h16*)alloc((size_t)R1p * 384 * 2);   // padded rows for tiled GEMM
  h16*   f2t   = (h16*)alloc((size_t)R1p * 256 * 2);
  h16*   h2    = (h16*)alloc((size_t)R1p * 128 * 2);
  int*   idx3b = (int*)alloc((size_t)R0 * 3 * 4);
  float* w3b   = (float*)alloc((size_t)R0 * 3 * 4);
  h16*   fp1A  = (h16*)alloc((size_t)R0 * 160 * 2);
  h16*   bigA  = (h16*)alloc((size_t)R0 * 128 * 2);
  h16*   bigB  = (h16*)alloc((size_t)R0 * 128 * 2);
  h16*   l3buf = (h16*)alloc((size_t)R0 * 16 * 2);

  const int TPB = 256;
  // input conversion + weight prep (idempotent, re-done every call)
  { int tot = R0 * 3; f2h_kernel<<<(tot + 255) / 256, TPB, 0, stream>>>(x, xh, tot); }
  for (int l = 0; l < 19; ++l) {
    const int tot = LDEFS[l].Np * LDEFS[l].Kp;
    prep_wt_kernel<<<(tot + 255) / 256, TPB, 0, stream>>>(
        P(LDEFS[l].wi), wt[l], LDEFS[l].Cin, LDEFS[l].Cout, LDEFS[l].Kp, LDEFS[l].Np);
  }
  // Generic GEMM launcher: 32x64 register-tiled WMMA kernel when Np % 64 == 0,
  // single-tile fallback otherwise (lin3). R padded to 64 rows (buffers sized).
  auto gemm = [&](const h16* A, int l, h16* O, int R, int act) {
    const int Np = LDEFS[l].Np;
    if ((Np & 63) == 0) {
      const int Rp = rpad64(R);
      const int tiles = (Rp >> 5) * (Np >> 6);
      gemm_tiled_kernel<<<(tiles + 7) / 8, TPB, 0, stream>>>(
          A, wt[l], P(LDEFS[l].bi), O, Rp, LDEFS[l].Kp, Np, LDEFS[l].Cout, act);
    } else {
      const int tiles = (R >> 4) * (Np >> 4);
      gemm_relu_kernel<<<(tiles + 7) / 8, TPB, 0, stream>>>(
          A, wt[l], P(LDEFS[l].bi), O, R, LDEFS[l].Kp, Np, LDEFS[l].Cout, act);
    }
  };

  // ---- SA1 ----
  fps_kernel<<<B_, 1024, 0, stream>>>(pos, N_, N1_, idx1);
  gather_pos_kernel<<<(R1 + 255) / 256, TPB, 0, stream>>>(pos, idx1, pos1, N1_, N_, R1);
  radius_knn_kernel<<<R1, TPB, 0, stream>>>(pos1, pos, N1_, N_, 0.2f * 0.2f, nidx1, ncnt1);
  sa_fused_kernel<32, 64, 64, 128, 3><<<R1, TPB, 0, stream>>>(
      pos, xh, pos1, nidx1, ncnt1,
      wt[13], P(LDEFS[13].bi), wt[14], P(LDEFS[14].bi), wt[15], P(LDEFS[15].bi), x1h);
  // ---- SA2 ----
  fps_kernel<<<B_, 1024, 0, stream>>>(pos1, N1_, N2_, idx2);
  gather_pos_kernel<<<(R2 + 255) / 256, TPB, 0, stream>>>(pos1, idx2, pos2, N2_, N1_, R2);
  radius_knn_kernel<<<R2, TPB, 0, stream>>>(pos2, pos1, N2_, N1_, 0.4f * 0.4f, nidx2, ncnt2);
  sa_fused_kernel<160, 128, 128, 256, 128><<<R2, TPB, 0, stream>>>(
      pos1, x1h, pos2, nidx2, ncnt2,
      wt[16], P(LDEFS[16].bi), wt[17], P(LDEFS[17].bi), wt[18], P(LDEFS[18].bi), x2h);
  // ---- Global SA ----
  { int tot = R2 * 288; build_gsaA_kernel<<<(tot + 255) / 256, TPB, 0, stream>>>(x2h, pos2, gsaA, tot); }
  gemm(gsaA, 7, gt1, R2, 1);
  gemm(gt1, 8, gt2, R2, 1);
  gemm(gt2, 9, gt3, R2, 0);
  gmax_kernel<<<B_, TPB, 0, stream>>>(gt3, gpool);
  // ---- FP3 (k=1 interp == broadcast of global feature) ----
  { int tot = R2 * 1280; build_fp3A_kernel<<<(tot + 255) / 256, TPB, 0, stream>>>(gpool, x2h, fp3A, tot); }
  gemm(fp3A, 5, f3t, R2, 1);
  gemm(f3t, 6, h3, R2, 0);
  // ---- FP2 ----
  knn3_kernel<<<(R1 + 255) / 256, TPB, 0, stream>>>(pos1, pos2, R1, N1_, N2_, idx3a, w3a);
  { int tot = R1 * 384; build_fp2A_kernel<<<(tot + 255) / 256, TPB, 0, stream>>>(h3, idx3a, w3a, x1h, fp2A, tot); }
  gemm(fp2A, 3, f2t, R1, 1);
  gemm(f2t, 4, h2, R1, 0);
  // ---- FP1 ----
  knn3_kernel<<<(R0 + 255) / 256, TPB, 0, stream>>>(pos, pos1, R0, N_, N1_, idx3b, w3b);
  { int tot = R0 * 160; build_fp1A_kernel<<<(tot + 255) / 256, TPB, 0, stream>>>(h2, idx3b, w3b, xh, fp1A, tot); }
  gemm(fp1A, 0, bigA, R0, 1);
  gemm(bigA, 1, bigB, R0, 1);
  gemm(bigB, 2, bigA, R0, 0);
  // ---- lin1/lin2 (leaky relu after linear), lin3 (linear) ----
  gemm(bigA, 10, bigB, R0, 2);
  gemm(bigB, 11, bigA, R0, 2);
  gemm(bigA, 12, l3buf, R0, 0);
  out_kernel<<<(R0 * 3 + 255) / 256, TPB, 0, stream>>>(l3buf, (float*)d_out, R0 * 3);
}
